// NeuralRootFind_48816598286979
// MI455X (gfx1250) — compile-verified, matrix operand order checked by reading, not run on hardware
//
#include <hip/hip_runtime.h>
#include <cstdint>
#include <cstddef>

typedef __attribute__((ext_vector_type(16))) _Float16 v16h;
typedef __attribute__((ext_vector_type(8)))  float    v8f;
typedef _Float16 f16;

#define DEV __device__ __forceinline__

// ---------------------------------------------------------------------------
// dims
// ---------------------------------------------------------------------------
// BATCH=16, C=64, NOISE=4, NX=NT=128, M1=M2=64, N_ITER=6
// flattened: bi/bj in [0,1024) = b*64 + channel ; point xt in [0,16384)

// ---------------------------------------------------------------------------
// helpers
// ---------------------------------------------------------------------------
DEV v8f vzero() {
  v8f z = {0.f, 0.f, 0.f, 0.f, 0.f, 0.f, 0.f, 0.f};
  return z;
}

DEV v8f wmma_f16(v16h a, v16h b, v8f c) {
  return __builtin_amdgcn_wmma_f32_16x16x32_f16(false, a, false, b,
                                                (short)0, c, false, false);
}

// Gather a 16x32 f16 WMMA operand fragment from an LDS row (row-major in K,
// K-pairs dword-contiguous).  K element for slot (v,j):
//   k = ((v&4)<<2) + (half<<3) + ((v&3)<<1) + j
DEV v16h ld_frag(const f16* rowbase, int half) {
  union U { v16h vec; unsigned u[8]; } f;
#pragma unroll
  for (int s = 0; s < 8; ++s) {
    int kb = ((s & 4) << 2) + (half << 3) + ((s & 3) << 1);
    f.u[s] = *reinterpret_cast<const unsigned*>(rowbase + kb);
  }
  return f.vec;
}

// CDNA5 async global->LDS copy, 16B per lane (ASYNCcnt-tracked, no VGPR
// round-trip).  lds_off = byte offset of destination within wave's LDS
// (flat-address low 32 bits), voff = per-lane byte offset added to saddr.
DEV void async_copy_b128(unsigned lds_off, unsigned voff, const void* saddr) {
  asm volatile("global_load_async_to_lds_b128 %0, %1, %2"
               :
               : "v"(lds_off), "v"(voff), "s"(saddr)
               : "memory");
}
DEV void async_wait0() {
  asm volatile("s_wait_asynccnt 0x0" ::: "memory");
}

// ---------------------------------------------------------------------------
// DFT matrix builders (all stored transposed: [n][k], f16)
//
// forward (fft + fftshift + truncate to 64 central modes), freq = tm-32:
//   F[tm,t] = exp(-2*pi*i*(tm-32)*t/128)
// inverse (ifftshift + zero-pad + ifft, 1/128 norm), freq = tm-32:
//   G[t,tm] = (1/128) exp(+2*pi*i*(tm-32)*t/128)
// init-time inverse (w was ifftshift'ed along M2 then zero-padded to 128):
//   Gi[t,tm] = (1/128) exp(+2*pi*i*((tm+32)%64)*t/128)
// complex contraction as real GEMM (K doubled):  A rows k=2idx+e (e=0:re,1:im)
//   B[2idx]   = ( br | bi ),  B[2idx+1] = ( -bi | br )
// ---------------------------------------------------------------------------
__global__ void build_dft(f16* out, int which) {
  int idx = blockIdx.x * 256 + threadIdx.x;
  const float PI2 = 6.283185307179586f;
  float val = 0.f;
  if (which == 0) {                 // Bt_fwd_T [n'=e'*64+tm][t=k], 128x128
    if (idx >= 128 * 128) return;
    int n = idx >> 7, t = idx & 127;
    int ep = n >> 6, tm = n & 63;
    float th = PI2 * (float)((tm - 32) * t) / 128.f;
    val = ep ? -__sinf(th) : __cosf(th);
  } else if (which == 1) {          // Bx_fwd_T [n'=e'*64+xm][k=2x+e], 128x256
    if (idx >= 128 * 256) return;
    int n = idx >> 8, k = idx & 255;
    int ep = n >> 6, xm = n & 63;
    int x = k >> 1, e = k & 1;
    float ph = PI2 * (float)((xm - 32) * x) / 128.f;
    float cr = __cosf(ph), ci = -__sinf(ph);
    val = ep ? (e ? cr : ci) : (e ? -ci : cr);
  } else if (which == 2 || which == 3) {  // Bt_inv(_init)_T [n'=e'*128+t][k=2tm+e]
    if (idx >= 256 * 128) return;
    int n = idx >> 7, k = idx & 127;
    int ep = n >> 7, t = n & 127;
    int tm = k >> 1, e = k & 1;
    float ps = (which == 2)
                   ? PI2 * (float)((tm - 32) * t) / 128.f
                   : PI2 * (float)((((tm + 32) & 63)) * t) / 128.f;
    float gr = __cosf(ps) * (1.f / 128.f), gi = __sinf(ps) * (1.f / 128.f);
    val = ep ? (e ? gr : gi) : (e ? -gi : gr);
  } else {                          // Bx_inv_T [n=x][k=2xm+e], real out only
    if (idx >= 128 * 128) return;
    int x = idx >> 7, k = idx & 127;
    int xm = k >> 1, e = k & 1;
    float ph = PI2 * (float)((xm - 32) * x) / 128.f;
    val = (e ? -__sinf(ph) : __cosf(ph)) * (1.f / 128.f);
  }
  out[idx] = (f16)val;
}

// Rearranged spectral weights: W[mode=xm*64+tm][n=ep*64+j][k=2i+e]  (f16)
// row 2i:   ( wr | wi )    row 2i+1: ( -wi | wr )   -- stored transposed [n][k]
__global__ void build_W(const float* __restrict__ wr,
                        const float* __restrict__ wi, f16* __restrict__ W) {
  size_t idx = (size_t)blockIdx.x * 256 + threadIdx.x;
  if (idx >= (size_t)4096 * 16384) return;
  int mode = (int)(idx >> 14);
  int r = (int)(idx & 16383);
  int n = r >> 7, k = r & 127;
  int xm = mode >> 6, tm = mode & 63;
  int i = k >> 1, e = k & 1, ep = n >> 6, j = n & 63;
  size_t src = (((size_t)i * 64 + j) * 64 + xm) * 64 + tm;
  float a = wr[src], b = wi[src];
  W[idx] = (f16)(e ? (ep ? a : -b) : (ep ? b : a));
}

// f16 copies of A (64x64, [o][i]) and Bw ([n][o][i])
__global__ void build_wts(const float* __restrict__ A,
                          const float* __restrict__ Bw, f16* Af16, f16* Bwf16) {
  int idx = blockIdx.x * 256 + threadIdx.x;
  if (idx < 4096) Af16[idx] = (f16)A[idx];
  if (idx < 4 * 4096) {
    int n = idx >> 12, o = (idx >> 6) & 63, i = idx & 63;
    Bwf16[idx] = (f16)Bw[(o * 4 + n) * 64 + i];  // Bw layout (C, NOISE, C)
  }
}

// broadcast Zhat (x-spectrum of z0) over the tm axis: sf[e][bi][xm][tm]
__global__ void bcast_sf(const f16* __restrict__ Zhat, f16* __restrict__ sf) {
  size_t idx = (size_t)blockIdx.x * 256 + threadIdx.x;
  if (idx >= (size_t)2 * 1024 * 4096) return;
  int xm = (int)((idx >> 6) & 63);
  int bi = (int)((idx >> 12) & 1023);
  int e = (int)(idx >> 22);
  sf[idx] = Zhat[bi * 128 + e * 64 + xm];
}

// ---------------------------------------------------------------------------
// generic WMMA GEMM stage:  C[row, n] = sum_k A(row,k) * B[n][k]
// workgroup: 4 waves, tile M=64 x N=128, K-step 32
// ---------------------------------------------------------------------------
enum { S_INIT1 = 0, S_FWD_T, S_FWD_X, S_INV_T, S_INV_X };

template <int STAGE>
DEV f16 load_a(const void* Ap, int row, int k) {
  if constexpr (STAGE == S_INIT1) {          // z0 f32 [bi][x]
    return (f16)((const float*)Ap)[(size_t)row * 128 + k];
  } else if constexpr (STAGE == S_FWD_T) {   // s f16 [bi*128+x][t]
    return ((const f16*)Ap)[(size_t)row * 128 + k];
  } else if constexpr (STAGE == S_FWD_X) {   // tmp f16 [bi][x][e*64+tm]
    int bi = row >> 6, tm = row & 63, x = k >> 1, e = k & 1;
    return ((const f16*)Ap)[((size_t)bi * 128 + x) * 128 + e * 64 + tm];
  } else if constexpr (STAGE == S_INV_T) {   // core f16 [e][bj][xm][tm]
    int bj = row >> 6, xm = row & 63, tm = k >> 1, e = k & 1;
    return ((const f16*)Ap)[(((size_t)e * 1024 + bj) * 64 + xm) * 64 + tm];
  } else {                                   // q f16 [e][bj][xm][t]
    int bj = row >> 7, t = row & 127, xm = k >> 1, e = k & 1;
    return ((const f16*)Ap)[(((size_t)e * 1024 + bj) * 64 + xm) * 128 + t];
  }
}

template <int STAGE>
DEV void store_c(void* Cp, const float* add, int row, int col, float val) {
  if constexpr (STAGE == S_INIT1 || STAGE == S_FWD_T) {
    ((f16*)Cp)[(size_t)row * 128 + col] = (f16)val;
  } else if constexpr (STAGE == S_FWD_X) {   // sf [e'][bi][xm][tm]
    int bi = row >> 6, tm = row & 63, ep = col >> 6, xm = col & 63;
    ((f16*)Cp)[(((size_t)ep * 1024 + bi) * 64 + xm) * 64 + tm] = (f16)val;
  } else if constexpr (STAGE == S_INV_T) {   // q [e'][bj][xm][t]
    int bj = row >> 6, xm = row & 63, ep = col >> 7, t = col & 127;
    ((f16*)Cp)[(((size_t)ep * 1024 + bj) * 64 + xm) * 128 + t] = (f16)val;
  } else {                                   // z f32 [bj][x][t] = z0_path + conv
    int bj = row >> 7, t = row & 127;
    size_t o = ((size_t)bj * 128 + col) * 128 + t;
    ((float*)Cp)[o] = (add ? add[o] : 0.f) + val;
  }
}

template <int STAGE>
__global__ __launch_bounds__(128) void gemm_stage(const void* __restrict__ Ap,
                                                  const f16* __restrict__ Bmat,
                                                  void* __restrict__ Cp,
                                                  const float* __restrict__ add) {
  constexpr int KT = (STAGE == S_FWD_X) ? 256 : 128;
  __shared__ f16 lA[64 * 32];   // [rowLocal][k]
  __shared__ f16 lB[128 * 32];  // [nLocal][k]
  const int tid = threadIdx.x, wave = tid >> 5, lane = tid & 31;
  const int m = lane & 15, half = lane >> 4;
  const int row0 = blockIdx.x * 64;
  const int n0 = blockIdx.y * 128;
  v8f acc[8];
#pragma unroll
  for (int nt = 0; nt < 8; ++nt) acc[nt] = vzero();

  for (int k0 = 0; k0 < KT; k0 += 32) {
#pragma unroll
    for (int it = 0; it < 16; ++it) {               // 64x32 A tile
      int idx = it * 128 + tid;
      lA[idx] = load_a<STAGE>(Ap, row0 + (idx >> 5), k0 + (idx & 31));
    }
    {                                               // 128x32 B tile (copy)
      const f16* src = Bmat + (size_t)(n0 + tid) * KT + k0;
      f16* dst = lB + tid * 32;
#pragma unroll
      for (int q2 = 0; q2 < 4; ++q2)
        reinterpret_cast<uint4*>(dst)[q2] =
            reinterpret_cast<const uint4*>(src)[q2];
    }
    __syncthreads();
    v16h af = ld_frag(lA + (wave * 16 + m) * 32, half);
#pragma unroll
    for (int nt = 0; nt < 8; ++nt) {
      v16h bf = ld_frag(lB + (nt * 16 + m) * 32, half);
      acc[nt] = wmma_f16(af, bf, acc[nt]);
    }
    __syncthreads();
  }
#pragma unroll
  for (int nt = 0; nt < 8; ++nt)
#pragma unroll
    for (int r = 0; r < 8; ++r)
      store_c<STAGE>(Cp, add, row0 + wave * 16 + r + 8 * half,
                     n0 + nt * 16 + m, acc[nt][r]);
}

// ---------------------------------------------------------------------------
// per-mode spectral channel mix: core[b, e'*64+j] = sf[b, 2i+e] @ W[mode]
// one workgroup per mode; M=16 (batch), K=128, N=128.
// W block (32KB, contiguous) is streamed cache->LDS with CDNA5 async loads
// (ASYNCcnt) while the A tile is gathered; next mode's block is prefetched.
// ---------------------------------------------------------------------------
__global__ __launch_bounds__(128) void mode_gemm(const f16* __restrict__ sf,
                                                 const f16* __restrict__ W,
                                                 f16* __restrict__ core,
                                                 int nmodes) {
  const int mode = blockIdx.x;
  const int xm = mode >> 6, tm = mode & 63;
  __shared__ f16 lA[16 * 128];    // [b][k]
  __shared__ f16 lB[128 * 128];   // [n][k]
  const int tid = threadIdx.x, wave = tid >> 5, lane = tid & 31;
  const int m = lane & 15, half = lane >> 4;

  // kick off async W stream first: 16 x b128 per lane, no VGPR round-trip
  {
    const void* wsrc = (const void*)(W + (size_t)mode * 16384);
    unsigned lds0 = (unsigned)(uintptr_t)(&lB[0]);
#pragma unroll
    for (int it = 0; it < 16; ++it) {
      unsigned off = (unsigned)((it * 128 + tid) * 16);
      async_copy_b128(lds0 + off, off, wsrc);
    }
    // stream next workgroup's W block toward L2 while this one computes
    if (mode + 1 < nmodes)
      __builtin_prefetch(W + (size_t)(mode + 1) * 16384 + tid * 128, 0, 1);
  }
  // gather A tile (strided f16) while the async stream is in flight
#pragma unroll
  for (int it = 0; it < 16; ++it) {
    int idx = it * 128 + tid;
    int b = idx >> 7, k = idx & 127, e = k & 1, i = k >> 1;
    lA[idx] = sf[(((size_t)e * 1024 + b * 64 + i) * 64 + xm) * 64 + tm];
  }
  async_wait0();
  __syncthreads();

  v8f acc[2] = {vzero(), vzero()};
#pragma unroll
  for (int ks = 0; ks < 4; ++ks) {
    v16h af = ld_frag(lA + m * 128 + ks * 32, half);
#pragma unroll
    for (int nt = 0; nt < 2; ++nt) {
      v16h bf = ld_frag(lB + (wave * 32 + nt * 16 + m) * 128 + ks * 32, half);
      acc[nt] = wmma_f16(af, bf, acc[nt]);
    }
  }
#pragma unroll
  for (int nt = 0; nt < 2; ++nt)
#pragma unroll
    for (int r = 0; r < 8; ++r) {
      int b = r + 8 * half;
      int n = wave * 32 + nt * 16 + m;
      int ep = n >> 6, j = n & 63;
      core[(((size_t)ep * 1024 + b * 64 + j) * 64 + xm) * 64 + tm] =
          (f16)acc[nt][r];
    }
}

// ---------------------------------------------------------------------------
// pointwise: s = tanh(A z + bA) + sum_n xi_n * (Bw_n z)   (per point)
// workgroup = one batch b, 64-point tile; M=64 channels, K=64, N=64
// ---------------------------------------------------------------------------
DEV void gemm64(const f16* lWm, const f16* lZm, int wave, int m, int half,
                v8f acc[4]) {
#pragma unroll
  for (int ks = 0; ks < 2; ++ks) {
    v16h af = ld_frag(lWm + (wave * 16 + m) * 64 + ks * 32, half);
#pragma unroll
    for (int nt = 0; nt < 4; ++nt) {
      v16h bf = ld_frag(lZm + (nt * 16 + m) * 64 + ks * 32, half);
      acc[nt] = wmma_f16(af, bf, acc[nt]);
    }
  }
}

__global__ __launch_bounds__(128) void pointwise(const float* __restrict__ z,
                                                 const float* __restrict__ xi,
                                                 const f16* __restrict__ Af16,
                                                 const f16* __restrict__ Bwf16,
                                                 const float* __restrict__ bA,
                                                 f16* __restrict__ sOut) {
  const int blk = blockIdx.x;
  const int b = blk >> 8, xt0 = (blk & 255) * 64;
  __shared__ f16 lZ[64 * 64];  // [point][i]
  __shared__ f16 lW[64 * 64];  // [o][i]
  const int tid = threadIdx.x, wave = tid >> 5, lane = tid & 31;
  const int m = lane & 15, half = lane >> 4;
#pragma unroll
  for (int it = 0; it < 32; ++it) {
    int idx = it * 128 + tid;
    int i = idx >> 6, c = idx & 63;
    lZ[c * 64 + i] = (f16)z[((size_t)b * 64 + i) * 16384 + xt0 + c];
    lW[idx] = Af16[idx];
  }
  __syncthreads();

  v8f acc[4];
#pragma unroll
  for (int nt = 0; nt < 4; ++nt) acc[nt] = vzero();
  gemm64(lW, lZ, wave, m, half, acc);

  float bav[8];
#pragma unroll
  for (int r = 0; r < 8; ++r) bav[r] = bA[wave * 16 + r + 8 * half];
  float sreg[4][8];
#pragma unroll
  for (int nt = 0; nt < 4; ++nt)
#pragma unroll
    for (int r = 0; r < 8; ++r) sreg[nt][r] = tanhf(acc[nt][r] + bav[r]);

  for (int n = 0; n < 4; ++n) {
    __syncthreads();
#pragma unroll
    for (int it = 0; it < 32; ++it)
      lW[it * 128 + tid] = Bwf16[n * 4096 + it * 128 + tid];
    __syncthreads();
#pragma unroll
    for (int nt = 0; nt < 4; ++nt) acc[nt] = vzero();
    gemm64(lW, lZ, wave, m, half, acc);
#pragma unroll
    for (int nt = 0; nt < 4; ++nt) {
      float xiv = xi[((size_t)b * 4 + n) * 16384 + xt0 + nt * 16 + m];
#pragma unroll
      for (int r = 0; r < 8; ++r) sreg[nt][r] += xiv * acc[nt][r];
    }
  }
#pragma unroll
  for (int nt = 0; nt < 4; ++nt)
#pragma unroll
    for (int r = 0; r < 8; ++r)
      sOut[((size_t)b * 64 + wave * 16 + r + 8 * half) * 16384 + xt0 +
           nt * 16 + m] = (f16)sreg[nt][r];
}

// ---------------------------------------------------------------------------
// host orchestration
// ---------------------------------------------------------------------------
extern "C" void kernel_launch(void* const* d_in, const int* in_sizes, int n_in,
                              void* d_out, int out_size, void* d_ws,
                              size_t ws_size, hipStream_t stream) {
  (void)in_sizes; (void)n_in; (void)out_size; (void)ws_size;
  const float* z0 = (const float*)d_in[0];
  const float* xi = (const float*)d_in[1];
  const float* wr = (const float*)d_in[2];
  const float* wi = (const float*)d_in[3];
  const float* A  = (const float*)d_in[4];
  const float* bA = (const float*)d_in[5];
  const float* Bw = (const float*)d_in[6];
  float* out = (float*)d_out;
  char* ws = (char*)d_ws;

  size_t off = 0;
  auto take = [&](size_t bytes) {
    size_t p = off;
    off += (bytes + 255) & ~(size_t)255;
    return p;
  };
  f16* W     = (f16*)(ws + take((size_t)4096 * 16384 * 2));  // 134 MB
  f16* BtF   = (f16*)(ws + take(128 * 128 * 2));
  f16* BxF   = (f16*)(ws + take(128 * 256 * 2));
  f16* BtI   = (f16*)(ws + take(256 * 128 * 2));
  f16* BtII  = (f16*)(ws + take(256 * 128 * 2));
  f16* BxI   = (f16*)(ws + take(128 * 128 * 2));
  f16* Af16  = (f16*)(ws + take(4096 * 2));
  f16* Bwf16 = (f16*)(ws + take(4 * 4096 * 2));
  f16* Zhat  = (f16*)(ws + take(1024 * 128 * 2));
  f16* sbuf  = (f16*)(ws + take((size_t)131072 * 128 * 2)); // 33.5 MB
  f16* tmpb  = (f16*)(ws + take((size_t)131072 * 128 * 2));
  f16* sfb   = (f16*)(ws + take((size_t)2 * 1024 * 4096 * 2)); // 16.8 MB
  f16* coreb = (f16*)(ws + take((size_t)2 * 1024 * 4096 * 2));
  f16* qb    = (f16*)(ws + take((size_t)2 * 1024 * 64 * 128 * 2)); // 33.5 MB
  float* z0p = (float*)(ws + take((size_t)1024 * 16384 * 4));      // 67 MB

  // precompute DFT matrices + f16 weights
  build_dft<<<64, 256, 0, stream>>>(BtF, 0);
  build_dft<<<128, 256, 0, stream>>>(BxF, 1);
  build_dft<<<128, 256, 0, stream>>>(BtI, 2);
  build_dft<<<128, 256, 0, stream>>>(BtII, 3);
  build_dft<<<64, 256, 0, stream>>>(BxI, 4);
  build_W<<<262144, 256, 0, stream>>>(wr, wi, W);
  build_wts<<<64, 256, 0, stream>>>(A, Bw, Af16, Bwf16);

  // init: z0_path = Re( iDFT_x( iDFT_t_pad( W . DFT_x(z0) ) ) )
  gemm_stage<S_INIT1><<<dim3(16, 1), 128, 0, stream>>>(z0, BtF, Zhat, nullptr);
  bcast_sf<<<32768, 256, 0, stream>>>(Zhat, sfb);
  mode_gemm<<<4096, 128, 0, stream>>>(sfb, W, coreb, 4096);
  gemm_stage<S_INV_T><<<dim3(1024, 2), 128, 0, stream>>>(coreb, BtII, qb, nullptr);
  gemm_stage<S_INV_X><<<dim3(2048, 1), 128, 0, stream>>>(qb, BxI, z0p, nullptr);
  hipMemcpyAsync(out, z0p, (size_t)1024 * 16384 * 4, hipMemcpyDeviceToDevice,
                 stream);

  // Picard iterations: z <- z0_path + S*(F(z) + G(z) xi)
  for (int it = 0; it < 6; ++it) {
    pointwise<<<4096, 128, 0, stream>>>(out, xi, Af16, Bwf16, bA, sbuf);
    gemm_stage<S_FWD_T><<<dim3(2048, 1), 128, 0, stream>>>(sbuf, BtF, tmpb, nullptr);
    gemm_stage<S_FWD_X><<<dim3(1024, 1), 128, 0, stream>>>(tmpb, BxF, sfb, nullptr);
    mode_gemm<<<4096, 128, 0, stream>>>(sfb, W, coreb, 4096);
    gemm_stage<S_INV_T><<<dim3(1024, 2), 128, 0, stream>>>(coreb, BtI, qb, nullptr);
    gemm_stage<S_INV_X><<<dim3(2048, 1), 128, 0, stream>>>(qb, BxI, out, z0p);
  }
}
